// Transformer_6107443495626
// MI455X (gfx1250) — matrix operand order repaired and checked by reading, not compiled, not run on hardware
//
#include <hip/hip_runtime.h>
#include <math.h>

// ---------------------------------------------------------------------------
// Types for CDNA5 WMMA (wave32)
// ---------------------------------------------------------------------------
typedef __attribute__((ext_vector_type(16))) __bf16 v16bf;
typedef __attribute__((ext_vector_type(8)))  float  v8f;

union FragBF {
    v16bf v;
    unsigned int u[8];
};

__device__ __forceinline__ unsigned short f2bf(float f) {
    unsigned int u = __float_as_uint(f);
    u += 0x7FFFu + ((u >> 16) & 1u);          // round-to-nearest-even
    return (unsigned short)(u >> 16);
}

// 16-lane butterfly step on the VALU (v_permlane16_b32) instead of LDS
// bpermute. Full 16-entry selector tables for xor 1/2/4/8.
#if __has_builtin(__builtin_amdgcn_permlane16)
__device__ __forceinline__ float pl16(float v, unsigned s0, unsigned s1) {
    unsigned r = __builtin_amdgcn_permlane16(
        __float_as_uint(v), __float_as_uint(v), s0, s1, false, false);
    return __uint_as_float(r);
}
__device__ __forceinline__ float rowmax16(float x) {
    x = fmaxf(x, pl16(x, 0x67452301u, 0xEFCDAB89u));   // xor 1
    x = fmaxf(x, pl16(x, 0x54761032u, 0xDCFE98BAu));   // xor 2
    x = fmaxf(x, pl16(x, 0x32107654u, 0xBA98FEDCu));   // xor 4
    x = fmaxf(x, pl16(x, 0xFEDCBA98u, 0x76543210u));   // xor 8
    return x;
}
#else
__device__ __forceinline__ float rowmax16(float x) {
    #pragma unroll
    for (int off = 8; off > 0; off >>= 1)
        x = fmaxf(x, __shfl_xor(x, off, 32));
    return x;
}
#endif

// ---------------------------------------------------------------------------
// Weight convert + transpose: fp32 W[K][N] -> bf16 Wt[N][K].
// 32x32 tiles through LDS; once per layer, off the GEMM critical path.
// ---------------------------------------------------------------------------
__global__ __launch_bounds__(256) void wcvtT_kernel(
    const float* __restrict__ w, unsigned short* __restrict__ o, int K, int N)
{
    __shared__ unsigned short t[32][33];
    const int tid = threadIdx.x;
    const int bn = blockIdx.x * 32;
    const int bk = blockIdx.y * 32;

    // load 32x32 fp32 tile, convert to bf16 in LDS
    {
        const int r = tid >> 3;             // 0..31 (k)
        const int c = (tid & 7) * 4;        // 0..28 (n)
        const float4 v = *(const float4*)(w + (size_t)(bk + r) * N + bn + c);
        t[r][c + 0] = f2bf(v.x);
        t[r][c + 1] = f2bf(v.y);
        t[r][c + 2] = f2bf(v.z);
        t[r][c + 3] = f2bf(v.w);
    }
    __syncthreads();
    // store transposed, coalesced 64-bit writes
    {
        const int rr = tid >> 3;            // 0..31 (n)
        const int cc = (tid & 7) * 4;       // 0..28 (k)
        unsigned int lo = (unsigned int)t[cc + 0][rr] |
                          ((unsigned int)t[cc + 1][rr] << 16);
        unsigned int hi = (unsigned int)t[cc + 2][rr] |
                          ((unsigned int)t[cc + 3][rr] << 16);
        ((uint2*)(o + (size_t)(bn + rr) * K + bk + cc))[0] = make_uint2(lo, hi);
    }
}

// ---------------------------------------------------------------------------
// LayerNorm: fp32 row (768) -> bf16 normalized row. One row per block.
// ---------------------------------------------------------------------------
__global__ __launch_bounds__(256) void ln_kernel(
    const float* __restrict__ x, const float* __restrict__ g,
    const float* __restrict__ bt, unsigned short* __restrict__ y)
{
    const int row = blockIdx.x;
    const int t = threadIdx.x;
    const float* xr = x + (size_t)row * 768;

    float v0 = xr[t], v1 = xr[t + 256], v2 = xr[t + 512];
    __shared__ float red[8];

    float s = v0 + v1 + v2;
    #pragma unroll
    for (int off = 16; off > 0; off >>= 1) s += __shfl_xor(s, off, 32);
    if ((t & 31) == 0) red[t >> 5] = s;
    __syncthreads();
    float tot = 0.f;
    #pragma unroll
    for (int i = 0; i < 8; i++) tot += red[i];
    const float mean = tot * (1.0f / 768.0f);
    __syncthreads();

    float c0 = v0 - mean, c1 = v1 - mean, c2 = v2 - mean;
    float q = c0 * c0 + c1 * c1 + c2 * c2;
    #pragma unroll
    for (int off = 16; off > 0; off >>= 1) q += __shfl_xor(q, off, 32);
    if ((t & 31) == 0) red[t >> 5] = q;
    __syncthreads();
    float vt = 0.f;
    #pragma unroll
    for (int i = 0; i < 8; i++) vt += red[i];
    const float inv = rsqrtf(vt * (1.0f / 768.0f) + 1e-5f);

    unsigned short* yr = y + (size_t)row * 768;
    yr[t]       = f2bf(c0 * inv * g[t]       + bt[t]);
    yr[t + 256] = f2bf(c1 * inv * g[t + 256] + bt[t + 256]);
    yr[t + 512] = f2bf(c2 * inv * g[t + 512] + bt[t + 512]);
}

// ---------------------------------------------------------------------------
// WMMA GEMM: C(MxN) = A(MxK, bf16) * Bt(NxK, bf16, pre-transposed) + bias
// Block: 256 threads (8 waves), tile 128x256, K-step 32, reg double-buffer.
// Wave grid 2x4: each wave -> 64x64 (4x4 tiles of 16x16), 16 WMMAs / K-step.
// All LDS staging is contiguous uint4 (ds_store_b128); rows padded to 40
// ushorts so the 16 rows a half-wave reads hit distinct bank groups.
// EPI: 0 = bf16 out (bias),  1 = fp32 residual in-place (+= acc+bias),
//      2 = bf16 out gelu(acc+bias)
// ---------------------------------------------------------------------------
template <int EPI>
__global__ __launch_bounds__(256) void gemm_kernel(
    const unsigned short* __restrict__ A,   // M x K bf16, row-major
    const unsigned short* __restrict__ Bt,  // N x K bf16 (transposed weights)
    const float* __restrict__ bias,         // N
    void* __restrict__ Out,                 // M x N (bf16 or fp32)
    int M, int N, int K)
{
    __shared__ __align__(16) unsigned short As[128][40];   // [m][k] (+pad)
    __shared__ __align__(16) unsigned short Bs[256][40];   // [n][k] (+pad)

    const int tid  = threadIdx.x;
    const int wave = tid >> 5;
    const int lane = tid & 31;
    const int half = lane >> 4;          // 0: lanes 0-15, 1: lanes 16-31
    const int l16  = lane & 15;

    const int wm = (wave >> 2) * 64;     // wave row offset in tile
    const int wn = (wave & 3) * 64;      // wave col offset in tile
    const int bm = blockIdx.y * 128;
    const int bn = blockIdx.x * 256;

    v8f acc[4][4];
    #pragma unroll
    for (int i = 0; i < 4; i++)
        #pragma unroll
        for (int j = 0; j < 4; j++)
            #pragma unroll
            for (int e = 0; e < 8; e++) acc[i][j][e] = 0.0f;

    const int akb = half * 8;            // A-frag K base per ISA layout
    const int bkb = half * 8;            // B-frag K base per ISA layout

    const int ar = tid >> 1;             // A staging: 16 contiguous bf16/thread
    const int ac = (tid & 1) * 16;
    const unsigned short* arow = A  + (size_t)(bm + ar) * K + ac;
    const unsigned short* brow = Bt + (size_t)(bn + tid) * K;  // one row/thread

    uint4 aReg[2], bReg[4];

    // prologue: stage K-tile 0 into registers
    {
        #pragma unroll
        for (int i = 0; i < 2; i++) aReg[i] = ((const uint4*)arow)[i];
        #pragma unroll
        for (int i = 0; i < 4; i++) bReg[i] = ((const uint4*)brow)[i];
    }

    for (int k0 = 0; k0 < K; k0 += 32) {
        // --- commit staged registers to LDS (pure b128 stores)
        {
            uint4* adst = (uint4*)&As[ar][ac];
            #pragma unroll
            for (int i = 0; i < 2; i++) adst[i] = aReg[i];
            uint4* bdst = (uint4*)&Bs[tid][0];
            #pragma unroll
            for (int i = 0; i < 4; i++) bdst[i] = bReg[i];
        }
        __syncthreads();

        // --- prefetch next K-tile into registers (global latency overlap)
        if (k0 + 32 < K) {
            #pragma unroll
            for (int i = 0; i < 2; i++)
                aReg[i] = ((const uint4*)(arow + k0 + 32))[i];
            #pragma unroll
            for (int i = 0; i < 4; i++)
                bReg[i] = ((const uint4*)(brow + k0 + 32))[i];
        }
        // speculative prefetch of the K+2 tile (global_prefetch_b8)
        if (k0 + 96 <= K) {
            __builtin_prefetch(brow + k0 + 64, 0, 1);
            __builtin_prefetch(arow + k0 + 64, 0, 1);
        }

        // --- fragments + 16 WMMAs
        FragBF af[4];
        #pragma unroll
        for (int mt = 0; mt < 4; mt++) {
            const int row = wm + mt * 16 + l16;
            #pragma unroll
            for (int r = 0; r < 8; r++) {
                const int kk = akb + ((r & 4) << 2) + 2 * (r & 3);
                af[mt].u[r] = *(const unsigned int*)&As[row][kk];
            }
        }
        #pragma unroll
        for (int nt = 0; nt < 4; nt++) {
            FragBF bfr;
            const int col = wn + nt * 16 + l16;
            #pragma unroll
            for (int r = 0; r < 8; r++)
                bfr.u[r] = *(const unsigned int*)&Bs[col][bkb + ((r & 4) << 2) + 2 * (r & 3)];
            #pragma unroll
            for (int mt = 0; mt < 4; mt++)
                acc[mt][nt] = __builtin_amdgcn_wmma_f32_16x16x32_bf16(
                    false, af[mt].v, false, bfr.v,
                    (short)0, acc[mt][nt], false, false);
        }
        __syncthreads();
    }

    // --- epilogue (C layout: lane half selects rows +8, VGPR r = row)
    #pragma unroll
    for (int mt = 0; mt < 4; mt++) {
        #pragma unroll
        for (int nt = 0; nt < 4; nt++) {
            const int cn = bn + wn + nt * 16 + l16;
            const float bv = bias[cn];
            #pragma unroll
            for (int r = 0; r < 8; r++) {
                const int cm = bm + wm + mt * 16 + half * 8 + r;
                const float v = acc[mt][nt][r] + bv;
                if (EPI == 0) {
                    ((unsigned short*)Out)[(size_t)cm * N + cn] = f2bf(v);
                } else if (EPI == 1) {
                    ((float*)Out)[(size_t)cm * N + cn] += v;   // residual
                } else {
                    const float gel = 0.5f * v * (1.0f + erff(v * 0.70710678118f));
                    ((unsigned short*)Out)[(size_t)cm * N + cn] = f2bf(gel);
                }
            }
        }
    }
}

// ---------------------------------------------------------------------------
// Flash attention: block = (b, h, 128-row q block). 8 waves, each owns
// 16 q rows. K/V staged in LDS per 32-key step; online softmax with
// permlane16 row-max and WMMA row-sum (P x ones); WMMA for S = Q K^T and
// O += P V.
// qkv: (B*N) x 2304 bf16  [q | k | v], each h*64+dh
// out: (B*N) x 768 bf16
// ---------------------------------------------------------------------------
__global__ __launch_bounds__(256) void attn_kernel(
    const unsigned short* __restrict__ qkv,
    unsigned short* __restrict__ out)
{
    __shared__ __align__(16) unsigned short Ks[32][72];     // [key][dh] (+pad)
    __shared__ __align__(16) unsigned short Vt[64][40];     // [dh][key] (+pad)
    __shared__ __align__(16) unsigned short Ps[8][16][40];  // per-wave P tile

    const int tid  = threadIdx.x;
    const int wave = tid >> 5;
    const int lane = tid & 31;
    const int half = lane >> 4;
    const int l16  = lane & 15;

    const int wg = blockIdx.x;
    const int qt = wg & 7;               // N/128 = 8 q blocks
    const int h  = (wg >> 3) % 12;
    const int b  = wg / 96;
    const int q0 = qt * 128 + wave * 16;
    const size_t RS = 2304;

    // persistent Q fragments (16 x 64 -> two 16x32 A-frags)
    FragBF qf[2];
    {
        const unsigned short* qp =
            qkv + (size_t)(b * 1024 + q0 + l16) * RS + h * 64;
        #pragma unroll
        for (int c = 0; c < 2; c++)
            #pragma unroll
            for (int r = 0; r < 8; r++) {
                const int kk = c * 32 + half * 8 + ((r & 4) << 2) + 2 * (r & 3);
                qf[c].u[r] = *(const unsigned int*)(qp + kk);
            }
    }
    // B-fragment of all-ones (bf16 1.0 packed) for WMMA row-sum
    FragBF ones;
    #pragma unroll
    for (int r = 0; r < 8; r++) ones.u[r] = 0x3F803F80u;

    float mrow[8], lrow[8], alpha[8];
    v8f o[4];
    #pragma unroll
    for (int r = 0; r < 8; r++) { mrow[r] = -1e30f; lrow[r] = 0.0f; }
    #pragma unroll
    for (int nt = 0; nt < 4; nt++)
        #pragma unroll
        for (int e = 0; e < 8; e++) o[nt][e] = 0.0f;

    const float scale = 0.125f;          // 1/sqrt(64)

    for (int kb = 0; kb < 32; kb++) {
        const int key0 = kb * 32;
        // stage K block (coalesced 8-elem runs, b128 LDS stores)
        {
            const int key = tid >> 3;
            const int d0 = (tid & 7) * 8;
            const uint4 v = *(const uint4*)
                (qkv + (size_t)(b * 1024 + key0 + key) * RS + 768 + h * 64 + d0);
            *(uint4*)&Ks[key][d0] = v;
        }
        // stage V block transposed: Vt[dh][key]
        {
            #pragma unroll
            for (int i = 0; i < 8; i++) {
                const int idx = tid * 8 + i;
                const int key = idx >> 6;
                const int dh = idx & 63;
                Vt[dh][key] =
                    qkv[(size_t)(b * 1024 + key0 + key) * RS + 1536 + h * 64 + dh];
            }
        }
        __syncthreads();

        // S = Q K^T  (16 q rows x 32 keys, two 16x16 WMMA tiles)
        v8f s[2];
        #pragma unroll
        for (int nt = 0; nt < 2; nt++) {
            #pragma unroll
            for (int e = 0; e < 8; e++) s[nt][e] = 0.0f;
            #pragma unroll
            for (int c = 0; c < 2; c++) {
                FragBF kf;
                const int keycol = nt * 16 + l16;
                #pragma unroll
                for (int r = 0; r < 8; r++) {
                    const int dh = c * 32 + half * 8 + ((r & 4) << 2) + 2 * (r & 3);
                    kf.u[r] = *(const unsigned int*)&Ks[keycol][dh];
                }
                s[nt] = __builtin_amdgcn_wmma_f32_16x16x32_bf16(
                    false, qf[c].v, false, kf.v, (short)0, s[nt], false, false);
            }
            #pragma unroll
            for (int e = 0; e < 8; e++) s[nt][e] *= scale;
        }

        // row max on VALU (permlane16 butterfly within 16-lane groups)
        #pragma unroll
        for (int r = 0; r < 8; r++) {
            const float mx = rowmax16(fmaxf(s[0][r], s[1][r]));
            const float mnew = fmaxf(mrow[r], mx);
            alpha[r] = __expf(mrow[r] - mnew);
            mrow[r] = mnew;
            s[0][r] = __expf(s[0][r] - mnew);
            s[1][r] = __expf(s[1][r] - mnew);
        }

        // relayout P (C layout) -> A fragment via per-wave LDS tile
        #pragma unroll
        for (int nt = 0; nt < 2; nt++)
            #pragma unroll
            for (int r = 0; r < 8; r++)
                Ps[wave][half * 8 + r][nt * 16 + l16] = f2bf(s[nt][r]);
        __syncthreads();

        FragBF pf;
        #pragma unroll
        for (int r = 0; r < 8; r++) {
            const int kk = half * 8 + ((r & 4) << 2) + 2 * (r & 3);
            pf.u[r] = *(const unsigned int*)&Ps[wave][l16][kk];
        }

        // row-sum via matrix unit: WMMA(P, ones) broadcasts row sums to all
        // lanes with the same C-layout row mapping as the O accumulators.
        v8f ssum;
        #pragma unroll
        for (int e = 0; e < 8; e++) ssum[e] = 0.0f;
        ssum = __builtin_amdgcn_wmma_f32_16x16x32_bf16(
            false, pf.v, false, ones.v, (short)0, ssum, false, false);
        #pragma unroll
        for (int r = 0; r < 8; r++)
            lrow[r] = lrow[r] * alpha[r] + ssum[r];

        // O = O*alpha + P V   (4 dh tiles of 16)
        #pragma unroll
        for (int nt = 0; nt < 4; nt++) {
            #pragma unroll
            for (int r = 0; r < 8; r++) o[nt][r] *= alpha[r];
            FragBF vf;
            const int dcol = nt * 16 + l16;
            #pragma unroll
            for (int r = 0; r < 8; r++) {
                const int kk = half * 8 + ((r & 4) << 2) + 2 * (r & 3);
                vf.u[r] = *(const unsigned int*)&Vt[dcol][kk];
            }
            o[nt] = __builtin_amdgcn_wmma_f32_16x16x32_bf16(
                false, pf.v, false, vf.v, (short)0, o[nt], false, false);
        }
        __syncthreads();
    }

    // finalize: divide by l, emit bf16
    #pragma unroll
    for (int nt = 0; nt < 4; nt++) {
        const int dh = nt * 16 + l16;
        #pragma unroll
        for (int r = 0; r < 8; r++) {
            const float v = o[nt][r] / lrow[r];
            out[(size_t)(b * 1024 + q0 + half * 8 + r) * 768 + h * 64 + dh] =
                f2bf(v);
        }
    }
}

// ---------------------------------------------------------------------------
// Host orchestration
// ---------------------------------------------------------------------------
extern "C" void kernel_launch(void* const* d_in, const int* in_sizes, int n_in,
                              void* d_out, int out_size, void* d_ws, size_t ws_size,
                              hipStream_t stream) {
    (void)in_sizes; (void)n_in; (void)out_size; (void)ws_size;

    const float* x     = (const float*)d_in[0];
    const float* ln1_g = (const float*)d_in[1];
    const float* ln1_b = (const float*)d_in[2];
    const float* Wqkv  = (const float*)d_in[3];
    const float* bqkv  = (const float*)d_in[4];
    const float* Wo    = (const float*)d_in[5];
    const float* bo    = (const float*)d_in[6];
    const float* ln2_g = (const float*)d_in[7];
    const float* ln2_b = (const float*)d_in[8];
    const float* W1    = (const float*)d_in[9];
    const float* b1    = (const float*)d_in[10];
    const float* W2    = (const float*)d_in[11];
    const float* b2    = (const float*)d_in[12];

    float* h = (float*)d_out;                       // residual stream, fp32
    hipMemcpyAsync(h, x, (size_t)8192 * 768 * sizeof(float),
                   hipMemcpyDeviceToDevice, stream);

    unsigned short* ws   = (unsigned short*)d_ws;   // bf16 scratch
    unsigned short* y    = ws;                           // 8192 x 768
    unsigned short* qkv  = y    + (size_t)8192 * 768;    // 8192 x 2304
    unsigned short* attn = qkv  + (size_t)8192 * 2304;   // 8192 x 768
    unsigned short* hid  = attn + (size_t)8192 * 768;    // 8192 x 3072
    unsigned short* wqT  = hid  + (size_t)8192 * 3072;   // 2304 x 768 (NxK)
    unsigned short* woT  = wqT  + (size_t)768 * 2304;    // 768 x 768
    unsigned short* w1T  = woT  + (size_t)768 * 768;     // 3072 x 768
    unsigned short* w2T  = w1T  + (size_t)768 * 3072;    // 768 x 3072

    for (int l = 0; l < 12; l++) {
        // per-layer bf16 weight convert+transpose (K x N -> N x K)
        wcvtT_kernel<<<dim3(72, 24), 256, 0, stream>>>(
            Wqkv + (size_t)l * 1769472, wqT, 768, 2304);
        wcvtT_kernel<<<dim3(24, 24), 256, 0, stream>>>(
            Wo   + (size_t)l * 589824,  woT, 768, 768);
        wcvtT_kernel<<<dim3(96, 24), 256, 0, stream>>>(
            W1   + (size_t)l * 2359296, w1T, 768, 3072);
        wcvtT_kernel<<<dim3(24, 96), 256, 0, stream>>>(
            W2   + (size_t)l * 2359296, w2T, 3072, 768);

        ln_kernel<<<8192, 256, 0, stream>>>(h, ln1_g + l * 768, ln1_b + l * 768, y);
        gemm_kernel<0><<<dim3(9, 64), 256, 0, stream>>>(
            y, wqT, bqkv + l * 2304, qkv, 8192, 2304, 768);
        attn_kernel<<<768, 256, 0, stream>>>(qkv, attn);
        gemm_kernel<1><<<dim3(3, 64), 256, 0, stream>>>(
            attn, woT, bo + l * 768, h, 8192, 768, 768);
        ln_kernel<<<8192, 256, 0, stream>>>(h, ln2_g + l * 768, ln2_b + l * 768, y);
        gemm_kernel<2><<<dim3(12, 64), 256, 0, stream>>>(
            y, w1T, b1 + l * 3072, hid, 8192, 3072, 768);
        gemm_kernel<1><<<dim3(3, 64), 256, 0, stream>>>(
            hid, w2T, b2 + l * 768, h, 8192, 768, 3072);
    }
}
